// MultiHeadAttention_16527034155682
// MI455X (gfx1250) — matrix operand order repaired and checked by reading, not compiled, not run on hardware
//
#include <hip/hip_runtime.h>

#define DM 512
#define HD 64
#define NH 8
#define SQ 2048
#define BB 4
#define SCP (SQ + 4)   // padded fp32 score row stride (banks rotate per row)
#define ABP (SQ + 8)   // padded bf16 attn row stride

typedef __bf16 bf16_t;
typedef __attribute__((ext_vector_type(16))) __bf16 v16bf;
typedef __attribute__((ext_vector_type(8)))  __bf16 v8bf;
typedef __attribute__((ext_vector_type(4)))  __bf16 v4bf;
typedef __attribute__((ext_vector_type(8)))  float  v8f;
typedef __attribute__((ext_vector_type(4)))  float  v4f;

union U16BF { v16bf v; __bf16 e[16]; v8bf h[2]; };
union U4BF  { v4bf v; __bf16 e[4]; };

// ---- fragment loaders (ISA 7.12.2 layouts, wave32) ----

// A 16x32 bf16: lane<16 -> row M=lane, K {k0..k0+7, k0+16..k0+23}
//               lane>=16 -> same row, K {k0+8..k0+15, k0+24..k0+31}
__device__ inline v16bf a_frag_bf16(const bf16_t* __restrict__ X, int ld, int m0, int k0, int lane) {
  int row = m0 + (lane & 15);
  int kb  = k0 + ((lane >> 4) << 3);
  const bf16_t* p = X + (long)row * ld + kb;
  U16BF a;
  a.h[0] = *(const v8bf*)p;        // K kb..kb+7
  a.h[1] = *(const v8bf*)(p + 16); // K kb+16..kb+23
  return a.v;
}

// B 32x16 bf16 (stored as Bt[n][k], k contiguous):
// lane<16 -> col N=lane, K {k0..k0+7, k0+16..k0+23} ; lane>=16 -> col N=lane-16, K {k0+8..k0+15, k0+24..k0+31}
__device__ inline v16bf b_frag_bf16(const bf16_t* __restrict__ Bt, int ld, int n0, int k0, int lane) {
  int n  = n0 + (lane & 15);
  int kb = k0 + ((lane >> 4) << 3);
  const bf16_t* p = Bt + (long)n * ld + kb;
  U16BF b;
  b.h[0] = *(const v8bf*)p;       // K kb..kb+7
  b.h[1] = *(const v8bf*)(p + 16); // K kb+16..kb+23
  return b.v;
}

__device__ inline v8f wmma_bf16(v16bf a, v16bf b, v8f c) {
  return __builtin_amdgcn_wmma_f32_16x16x32_bf16(false, a, false, b, (short)0, c, false, false);
}

// ---- kernel 0: fp32 -> bf16 conversion ----
__global__ void cvt_kernel(const float* __restrict__ src, bf16_t* __restrict__ dst, long n) {
  long i = (((long)blockIdx.x * blockDim.x) + threadIdx.x) << 3;
  if (i >= n) return;
  float4 f0 = *(const float4*)(src + i);
  float4 f1 = *(const float4*)(src + i + 4);
  U16BF o;
  o.e[0]=(__bf16)f0.x; o.e[1]=(__bf16)f0.y; o.e[2]=(__bf16)f0.z; o.e[3]=(__bf16)f0.w;
  o.e[4]=(__bf16)f1.x; o.e[5]=(__bf16)f1.y; o.e[6]=(__bf16)f1.z; o.e[7]=(__bf16)f1.w;
  *(v8bf*)(dst + i) = o.h[0];
}

// ---- kernel 1: Y = X @ W^T + bias (bf16 in, bf16 out), double-buffered K loop ----
__global__ void proj_kernel(const bf16_t* __restrict__ X, const bf16_t* __restrict__ W,
                            const float* __restrict__ bias, bf16_t* __restrict__ Y, int mode) {
  int lane = threadIdx.x & 31;
  int wid  = blockIdx.x * (blockDim.x >> 5) + (threadIdx.x >> 5);
  int m0 = (wid & 511) << 4;
  int n0 = (wid >> 9) << 6;
  v8f acc[4] = {};
  v16bf a_cur = a_frag_bf16(X, DM, m0, 0, lane);
  v16bf b_cur[4];
#pragma unroll
  for (int t = 0; t < 4; t++) b_cur[t] = b_frag_bf16(W, DM, n0 + t * 16, 0, lane);
  for (int ks = 1; ks < 16; ks++) {
    int kn = ks << 5;
    v16bf a_n = a_frag_bf16(X, DM, m0, kn, lane);
    v16bf b_n[4];
#pragma unroll
    for (int t = 0; t < 4; t++) b_n[t] = b_frag_bf16(W, DM, n0 + t * 16, kn, lane);
#pragma unroll
    for (int t = 0; t < 4; t++) acc[t] = wmma_bf16(a_cur, b_cur[t], acc[t]);
    a_cur = a_n;
#pragma unroll
    for (int t = 0; t < 4; t++) b_cur[t] = b_n[t];
  }
#pragma unroll
  for (int t = 0; t < 4; t++) acc[t] = wmma_bf16(a_cur, b_cur[t], acc[t]);

  int g = lane >> 4, cl = lane & 15;
#pragma unroll
  for (int t = 0; t < 4; t++) {
    int col = n0 + t * 16 + cl;
    float bv = bias[col];
    int h = col >> 6, d = col & 63;
#pragma unroll
    for (int r = 0; r < 8; r++) {
      int m = m0 + r + (g << 3);
      int bb = m >> 11, s = m & 2047;
      float y = acc[t][r] + bv;
      long idx = (mode == 2) ? (((long)(bb * NH + h) * HD + d) * SQ + s)
                             : (((long)(bb * NH + h) * SQ + s) * HD + d);
      Y[idx] = (bf16_t)y;
    }
  }
}

// ---- kernel 2: attention for one 16-query strip of one (b,h) ----
__global__ void attn_kernel(const bf16_t* __restrict__ qh, const bf16_t* __restrict__ kh,
                            const bf16_t* __restrict__ vt, bf16_t* __restrict__ obuf,
                            float* __restrict__ attn_out) {
  extern __shared__ char smem[];
  float*  sc = (float*)smem;                                 // 16*SCP fp32 @ 0
  bf16_t* ab = (bf16_t*)(smem + 16 * SCP * 4);               // 16*ABP bf16
  const unsigned ASY = 16 * SCP * 4 + 16 * ABP * 2;          // async V bufs: 4 waves*2*1KB
  int tid = threadIdx.x;
  int lane = tid & 31, w = tid >> 5;
  int bid = blockIdx.x;
  int q0 = (bid & 127) << 4;
  int h  = (bid >> 7) & 7;
  int b  = bid >> 10;
  const bf16_t* qbh = qh + (long)(b * NH + h) * SQ * HD;
  const bf16_t* kbh = kh + (long)(b * NH + h) * SQ * HD;
  const bf16_t* vbh = vt + (long)(b * NH + h) * HD * SQ;
  int g = lane >> 4, cl = lane & 15;

  // ---- phase 1: scaled scores: s' = (QK^T) * (0.125 * log2 e)  (exp2 domain) ----
  const float SCALE2 = 0.125f * 1.4426950408889634f;
  v16bf aq0 = a_frag_bf16(qbh, HD, q0, 0, lane);
  v16bf aq1 = a_frag_bf16(qbh, HD, q0, 32, lane);
  v16bf bk0 = b_frag_bf16(kbh, HD, (w << 9), 0, lane);
  v16bf bk1 = b_frag_bf16(kbh, HD, (w << 9), 32, lane);
  for (int nt = 0; nt < 32; nt++) {
    int n0 = (w << 9) + (nt << 4);
    v16bf nb0, nb1;
    if (nt < 31) {
      nb0 = b_frag_bf16(kbh, HD, n0 + 16, 0, lane);
      nb1 = b_frag_bf16(kbh, HD, n0 + 16, 32, lane);
      if (nt + 4 < 32)
        __builtin_prefetch(kbh + (long)(n0 + 64 + cl) * HD, 0, 1);
    }
    v8f accv = {};
    accv = wmma_bf16(aq0, bk0, accv);
    accv = wmma_bf16(aq1, bk1, accv);
#pragma unroll
    for (int r = 0; r < 8; r++)
      sc[(r + (g << 3)) * SCP + n0 + cl] = accv[r] * SCALE2;
    bk0 = nb0; bk1 = nb1;
  }
  __syncthreads();

  // ---- phase 2: row softmax, float4-vectorized (8 threads/row, 16B interleave) ----
  {
    int row = tid >> 3;
    int seg = tid & 7;                     // kx = seg*4 + j*32
    float* srow = sc + row * SCP;
    float4 m4 = make_float4(-3.0e38f, -3.0e38f, -3.0e38f, -3.0e38f);
    for (int j = 0; j < 64; j++) {
      float4 v4 = *(const float4*)(srow + (seg << 2) + (j << 5));
      m4.x = fmaxf(m4.x, v4.x); m4.y = fmaxf(m4.y, v4.y);
      m4.z = fmaxf(m4.z, v4.z); m4.w = fmaxf(m4.w, v4.w);
    }
    float m = fmaxf(fmaxf(m4.x, m4.y), fmaxf(m4.z, m4.w));
    m = fmaxf(m, __shfl_xor(m, 1, 32));
    m = fmaxf(m, __shfl_xor(m, 2, 32));
    m = fmaxf(m, __shfl_xor(m, 4, 32));
    float s = 0.f;
    for (int j = 0; j < 64; j++) {
      float* p = srow + (seg << 2) + (j << 5);
      float4 v4 = *(const float4*)p;
      float4 e4;
      e4.x = exp2f(v4.x - m); e4.y = exp2f(v4.y - m);
      e4.z = exp2f(v4.z - m); e4.w = exp2f(v4.w - m);
      *(float4*)p = e4;
      s += (e4.x + e4.y) + (e4.z + e4.w);
    }
    s += __shfl_xor(s, 1, 32);
    s += __shfl_xor(s, 2, 32);
    s += __shfl_xor(s, 4, 32);
    float inv = 1.0f / s;
    float*  arow  = attn_out + (((long)(b * NH + h) * SQ + q0 + row) * SQ);
    bf16_t* abrow = ab + row * ABP;
    for (int j = 0; j < 64; j++) {
      int kx = (seg << 2) + (j << 5);
      float4 e4 = *(const float4*)(srow + kx);
      v4f pv = { e4.x * inv, e4.y * inv, e4.z * inv, e4.w * inv };
      __builtin_nontemporal_store(pv, (v4f*)(arow + kx));  // 536MB stream: NT hint
      U4BF pk;
      pk.e[0] = (bf16_t)pv.x; pk.e[1] = (bf16_t)pv.y;
      pk.e[2] = (bf16_t)pv.z; pk.e[3] = (bf16_t)pv.w;
      *(v4bf*)(abrow + kx) = pk.v;
    }
  }
  __syncthreads();

  // ---- phase 3: O(16x64) = attn @ V; async-to-LDS staged V, double-buffered ----
  int d0 = w << 4;
  unsigned mybuf = ASY + (unsigned)w * 2048u;
  int rowd = lane >> 1;
  int half = (lane & 1) << 4;
  const char* gsrc = (const char*)vbh + ((long)(d0 + rowd) * SQ) * 2;

  auto issue_chunk = [&](int kt, int bufidx) {
    unsigned l0 = mybuf + (unsigned)bufidx * 1024u + (unsigned)rowd * 64u + (unsigned)half;
    unsigned long long g0 = (unsigned long long)gsrc + ((unsigned long long)(kt << 5) << 1) + (unsigned)half;
    unsigned l1 = l0 + 32u;
    unsigned long long g1 = g0 + 32ull;
    asm volatile("global_load_async_to_lds_b128 %0, %1, off" :: "v"(l0), "v"(g0) : "memory");
    asm volatile("global_load_async_to_lds_b128 %0, %1, off" :: "v"(l1), "v"(g1) : "memory");
  };

  v8f oc = {};
  issue_chunk(0, 0);
  for (int kt = 0; kt < 63; kt++) {
    issue_chunk(kt + 1, (kt + 1) & 1);
    asm volatile("s_wait_asynccnt 0x2" ::: "memory");
    const bf16_t* chunk = (const bf16_t*)(smem + mybuf + (unsigned)(kt & 1) * 1024u);
    v16bf pa = a_frag_bf16(ab, ABP, 0, kt << 5, lane);
    v16bf vb = b_frag_bf16(chunk, 32, 0, 0, lane);
    oc = wmma_bf16(pa, vb, oc);
  }
  asm volatile("s_wait_asynccnt 0x0" ::: "memory");
  {
    const bf16_t* chunk = (const bf16_t*)(smem + mybuf + (unsigned)(63 & 1) * 1024u);
    v16bf pa = a_frag_bf16(ab, ABP, 0, 63 << 5, lane);
    v16bf vb = b_frag_bf16(chunk, 32, 0, 0, lane);
    oc = wmma_bf16(pa, vb, oc);
  }
#pragma unroll
  for (int r = 0; r < 8; r++) {
    int srw = q0 + r + (g << 3);
    int col = h * HD + d0 + cl;
    obuf[((long)(b * SQ + srw)) * DM + col] = (bf16_t)oc[r];
  }
}

// ---- kernel 3: out = O @ Wo^T + bo (bf16 in, fp32 out), double-buffered ----
__global__ void oproj_kernel(const bf16_t* __restrict__ X, const bf16_t* __restrict__ W,
                             const float* __restrict__ bias, float* __restrict__ out) {
  int lane = threadIdx.x & 31;
  int wid  = blockIdx.x * (blockDim.x >> 5) + (threadIdx.x >> 5);
  int m0 = (wid & 511) << 4;
  int n0 = (wid >> 9) << 6;
  v8f acc[4] = {};
  v16bf a_cur = a_frag_bf16(X, DM, m0, 0, lane);
  v16bf b_cur[4];
#pragma unroll
  for (int t = 0; t < 4; t++) b_cur[t] = b_frag_bf16(W, DM, n0 + t * 16, 0, lane);
  for (int ks = 1; ks < 16; ks++) {
    int kn = ks << 5;
    v16bf a_n = a_frag_bf16(X, DM, m0, kn, lane);
    v16bf b_n[4];
#pragma unroll
    for (int t = 0; t < 4; t++) b_n[t] = b_frag_bf16(W, DM, n0 + t * 16, kn, lane);
#pragma unroll
    for (int t = 0; t < 4; t++) acc[t] = wmma_bf16(a_cur, b_cur[t], acc[t]);
    a_cur = a_n;
#pragma unroll
    for (int t = 0; t < 4; t++) b_cur[t] = b_n[t];
  }
#pragma unroll
  for (int t = 0; t < 4; t++) acc[t] = wmma_bf16(a_cur, b_cur[t], acc[t]);

  int g = lane >> 4, cl = lane & 15;
#pragma unroll
  for (int t = 0; t < 4; t++) {
    int col = n0 + t * 16 + cl;
    float bv = bias[col];
#pragma unroll
    for (int r = 0; r < 8; r++) {
      int m = m0 + r + (g << 3);
      out[(long)m * DM + col] = acc[t][r] + bv;
    }
  }
}

extern "C" void kernel_launch(void* const* d_in, const int* in_sizes, int n_in,
                              void* d_out, int out_size, void* d_ws, size_t ws_size,
                              hipStream_t stream) {
  const float* q   = (const float*)d_in[0];
  const float* k   = (const float*)d_in[1];
  const float* v   = (const float*)d_in[2];
  const float* w_q = (const float*)d_in[3];
  const float* b_q = (const float*)d_in[4];
  const float* w_k = (const float*)d_in[5];
  const float* b_k = (const float*)d_in[6];
  const float* w_v = (const float*)d_in[7];
  const float* b_v = (const float*)d_in[8];
  const float* w_o = (const float*)d_in[9];
  const float* b_o = (const float*)d_in[10];
  (void)w_o;

  float* out  = (float*)d_out;
  float* attn = out + (long)BB * SQ * DM;

  const long NX = (long)BB * SQ * DM;
  const long NW = (long)DM * DM;
  bf16_t* xq  = (bf16_t*)d_ws;
  bf16_t* xk  = xq + NX;
  bf16_t* xv  = xk + NX;
  bf16_t* wqb = xv + NX;
  bf16_t* wkb = wqb + NW;
  bf16_t* wvb = wkb + NW;
  bf16_t* wob = wvb + NW;
  bf16_t* qh  = wob + NW;
  bf16_t* kh  = qh + NX;
  bf16_t* vt  = kh + NX;     // [B,H,hd,S]
  bf16_t* ob  = vt + NX;     // [B*S, DM]

  dim3 blk(128);
  dim3 grid_gemm(1024);

  cvt_kernel<<<dim3((unsigned)(NX / 8 / 256)), dim3(256), 0, stream>>>(q, xq, NX);
  cvt_kernel<<<dim3((unsigned)(NX / 8 / 256)), dim3(256), 0, stream>>>(k, xk, NX);
  cvt_kernel<<<dim3((unsigned)(NX / 8 / 256)), dim3(256), 0, stream>>>(v, xv, NX);
  cvt_kernel<<<dim3((unsigned)(NW / 8 / 256)), dim3(256), 0, stream>>>(w_q, wqb, NW);
  cvt_kernel<<<dim3((unsigned)(NW / 8 / 256)), dim3(256), 0, stream>>>(w_k, wkb, NW);
  cvt_kernel<<<dim3((unsigned)(NW / 8 / 256)), dim3(256), 0, stream>>>(w_v, wvb, NW);
  cvt_kernel<<<dim3((unsigned)(NW / 8 / 256)), dim3(256), 0, stream>>>(w_o, wob, NW);

  proj_kernel<<<grid_gemm, blk, 0, stream>>>(xq, wqb, b_q, qh, 0);
  proj_kernel<<<grid_gemm, blk, 0, stream>>>(xk, wkb, b_k, kh, 1);
  proj_kernel<<<grid_gemm, blk, 0, stream>>>(xv, wvb, b_v, vt, 2);

  size_t lds_bytes = (size_t)16 * SCP * 4 + (size_t)16 * ABP * 2 + 4 * 2 * 1024;  // ~205 KB
  attn_kernel<<<dim3(BB * NH * (SQ / 16)), blk, lds_bytes, stream>>>(qh, kh, vt, ob, attn);

  oproj_kernel<<<grid_gemm, blk, 0, stream>>>(ob, wob, b_o, out);
}
